// Pointnet_Backbone_12584254177601
// MI455X (gfx1250) — matrix operand order repaired and checked by reading, not compiled, not run on hardware
//
#include <hip/hip_runtime.h>

typedef __attribute__((ext_vector_type(16))) _Float16 v16h;
typedef __attribute__((ext_vector_type(8)))  _Float16 v8h;
typedef __attribute__((ext_vector_type(8)))  float    v8f;

// ---------------- workspace layout (all offsets multiples of 256) ----------------
constexpr size_t OFF_XYZ0   = 0;                         // 4*8192*3 f32   = 393216 B
constexpr size_t OFF_NXYZ1  = OFF_XYZ0   + 393216;       // 4*2048*3 f32   =  98304 B
constexpr size_t OFF_NXYZ2  = OFF_NXYZ1  + 98304;        // 4*512*3  f32   =  24576 B
constexpr size_t OFF_NXYZ3  = OFF_NXYZ2  + 24576;        // 4*128*3  f32   =   6144 B
constexpr size_t OFF_FEATS0 = OFF_NXYZ3  + 6144;         // 4*8192*3 f16   = 196608 B
constexpr size_t OFF_FEATS1 = OFF_FEATS0 + 196608;       // 4*2048*128 f16 = 2097152 B
constexpr size_t OFF_FEATS2 = OFF_FEATS1 + 2097152;      // 4*512*256 f16  = 1048576 B
constexpr size_t OFF_FEATS3 = OFF_FEATS2 + 1048576;      // 4*128*256 f16  = 262144 B
constexpr size_t OFF_FIDX   = OFF_FEATS3 + 262144;       // 4*2048 i32     =  32768 B
constexpr size_t OFF_WF16   = OFF_FIDX   + 32768;        // 256*512 f16    = 262144 B
constexpr size_t OFF_BUFA   = OFF_WF16   + 262144;       // 32 MB
constexpr size_t OFF_BUFB   = OFF_BUFA   + 33554432;     // 64 MB
// total ~105 MB

// ---------------- split pointcloud into xyz (f32) + feats (f16) ----------------
__global__ void pn_split_kernel(const float* __restrict__ pc, float* __restrict__ xyz,
                                _Float16* __restrict__ feats, int total) {
  int tid = blockIdx.x * blockDim.x + threadIdx.x;
  if (tid >= total) return;
  const float* p = pc + (size_t)tid * 6;
  xyz[tid * 3 + 0] = p[0];
  xyz[tid * 3 + 1] = p[1];
  xyz[tid * 3 + 2] = p[2];
  feats[tid * 3 + 0] = (_Float16)p[3];
  feats[tid * 3 + 1] = (_Float16)p[4];
  feats[tid * 3 + 2] = (_Float16)p[5];
}

// ---------------- farthest point sampling: 1 block per batch, LDS-resident ----------------
#define FPS_THREADS 1024
__global__ __launch_bounds__(FPS_THREADS)
void pn_fps_kernel(const float* __restrict__ xyz, int* __restrict__ fidx,
                   int N, int S) {
  __shared__ float mind[8192];
  __shared__ float rmax[FPS_THREADS];
  __shared__ int   rarg[FPS_THREADS];
  __shared__ int   s_far;
  const int b = blockIdx.x;
  const int t = threadIdx.x;
  const float* px = xyz + (size_t)b * N * 3;
  for (int i = t; i < N; i += FPS_THREADS) mind[i] = 1e10f;
  if (t == 0) s_far = 0;
  __syncthreads();
  for (int it = 0; it < S; ++it) {
    const int far = s_far;
    if (t == 0) fidx[b * S + it] = far;
    const float cx = px[far * 3 + 0], cy = px[far * 3 + 1], cz = px[far * 3 + 2];
    float best = -1.0f; int bestI = 0;
    for (int i = t; i < N; i += FPS_THREADS) {
      float dx = px[i * 3 + 0] - cx, dy = px[i * 3 + 1] - cy, dz = px[i * 3 + 2] - cz;
      float m = fminf(mind[i], dx * dx + dy * dy + dz * dz);
      mind[i] = m;
      if (m > best) { best = m; bestI = i; }
    }
    rmax[t] = best; rarg[t] = bestI;
    __syncthreads();
    for (int off = FPS_THREADS >> 1; off > 0; off >>= 1) {
      if (t < off) {
        float o = rmax[t + off];
        if (o > rmax[t] || (o == rmax[t] && rarg[t + off] < rarg[t])) {
          rmax[t] = o; rarg[t] = rarg[t + off];
        }
      }
      __syncthreads();
    }
    if (t == 0) s_far = rarg[0];
    __syncthreads();
  }
}

// ---------------- gather sampled centers ----------------
__global__ void pn_gather_kernel(const float* __restrict__ xyz, const int* __restrict__ fidx,
                                 float* __restrict__ nxyz, int B, int N, int S) {
  int tid = blockIdx.x * blockDim.x + threadIdx.x;
  if (tid >= B * S) return;
  int b = tid / S;
  int idx = fidx[tid];
  const float* p = xyz + ((size_t)b * N + idx) * 3;
  nxyz[tid * 3 + 0] = p[0];
  nxyz[tid * 3 + 1] = p[1];
  nxyz[tid * 3 + 2] = p[2];
}

// ---------------- ball query + grouping: one wave32 per center ----------------
__global__ void pn_group_kernel(const float* __restrict__ xyz, const _Float16* __restrict__ feats,
                                const float* __restrict__ nxyz, _Float16* __restrict__ out,
                                int B, int N, int S, int C, int Kpad, float r2) {
  const int lane = threadIdx.x & 31;
  const int wave = threadIdx.x >> 5;
  const int center = blockIdx.x * (blockDim.x >> 5) + wave;
  if (center >= B * S) return;
  const int b = center / S;
  const float* px = xyz + (size_t)b * N * 3;
  const float cx = nxyz[center * 3 + 0];
  const float cy = nxyz[center * 3 + 1];
  const float cz = nxyz[center * 3 + 2];

  int t = 0, myIdx = -1, firstIdx = 0;
  for (int base = 0; base < N && t < 32; base += 32) {
    const int p = base + lane;                       // N is a multiple of 32
    const float dx = px[p * 3 + 0] - cx;
    const float dy = px[p * 3 + 1] - cy;
    const float dz = px[p * 3 + 2] - cz;
    const bool inR = (dx * dx + dy * dy + dz * dz) <= r2;
    const unsigned mask = (unsigned)__ballot(inR);   // wave32: low 32 bits
    const int cnt = __popc(mask);
    if (t == 0 && cnt > 0) firstIdx = base + __builtin_ctz(mask);
    const int want = lane - t;                       // lane takes the want-th set bit
    if (myIdx < 0 && want >= 0 && want < cnt) {
      unsigned m = mask;
      for (int q = 0; q < want; ++q) m &= (m - 1);
      myIdx = base + __builtin_ctz(m);
    }
    t += cnt;
  }
  if (myIdx < 0) myIdx = firstIdx;                   // pad with first in-radius index

  _Float16* row = out + ((size_t)center * 32 + lane) * Kpad;
  row[0] = (_Float16)(px[myIdx * 3 + 0] - cx);
  row[1] = (_Float16)(px[myIdx * 3 + 1] - cy);
  row[2] = (_Float16)(px[myIdx * 3 + 2] - cz);
  const _Float16* f = feats + ((size_t)b * N + myIdx) * C;
  for (int c = 0; c < C; ++c) row[3 + c] = f[c];
  for (int c = 3 + C; c < Kpad; ++c) row[c] = (_Float16)0.f;
}

// ---------------- convert weights (Cout x Cin f32) -> (Cout x Kpad f16, zero padded) ----------------
__global__ void pn_convw_kernel(const float* __restrict__ w, _Float16* __restrict__ wf,
                                int Cin, int Kpad, int total) {
  int tid = blockIdx.x * blockDim.x + threadIdx.x;
  if (tid >= total) return;
  int r = tid / Kpad, c = tid % Kpad;
  wf[tid] = (c < Cin) ? (_Float16)w[r * Cin + c] : (_Float16)0.f;
}

// ---------------- WMMA GEMM: D = A(MxK) * W(NxK)^T, fused bias(+affine+ReLU) ----------------
// Each wave computes a 32x32 output tile: 4 independent accumulators, 4 WMMAs per
// 8 b128 loads (16 flop/B from L2) and 4 independent WMMA dependency chains.
// mode 0: out = f16, relu((acc+bias)*gamma+beta)   mode 1: out = f32, acc+bias
__global__ __launch_bounds__(256)
void pn_wmma_gemm_kernel(const _Float16* __restrict__ A, const _Float16* __restrict__ W,
                         const float* __restrict__ bias, const float* __restrict__ gamma,
                         const float* __restrict__ beta,
                         _Float16* __restrict__ outH, float* __restrict__ outF,
                         int M, int Kpad, int Cout, int mode) {
  const int lane = threadIdx.x & 31;
  const int wave = threadIdx.x >> 5;
  const int tiles_n = Cout >> 5;                     // 32-wide N tiles
  const int tile = blockIdx.x * (blockDim.x >> 5) + wave;
  if (tile >= (M >> 5) * tiles_n) return;            // wave-uniform: EXEC stays all-1s
  const int m0 = (tile / tiles_n) << 5;
  const int n0 = (tile % tiles_n) << 5;
  const int lr = lane & 15;                          // A: row / B: output channel
  const int lh = lane >> 4;                          // K-half selector

  v8f acc00 = {}, acc01 = {}, acc10 = {}, acc11 = {};
  const _Float16* arow0 = A + (size_t)(m0 + lr)      * Kpad + lh * 8;
  const _Float16* arow1 = A + (size_t)(m0 + 16 + lr) * Kpad + lh * 8;
  const _Float16* brow0 = W + (size_t)(n0 + lr)      * Kpad + lh * 8;
  const _Float16* brow1 = W + (size_t)(n0 + 16 + lr) * Kpad + lh * 8;
  for (int kb = 0; kb < Kpad; kb += 32) {
    // A fragments (16x32 f16): VGPR0-3 <- K = kb + lh*8 + [0..7], VGPR4-7 <- +16
    v8h a0lo = *(const v8h*)(arow0 + kb);
    v8h a0hi = *(const v8h*)(arow0 + kb + 16);
    v8h a1lo = *(const v8h*)(arow1 + kb);
    v8h a1hi = *(const v8h*)(arow1 + kb + 16);
    v16h a0, a1;
#pragma unroll
    for (int i = 0; i < 8; ++i) {
      a0[i] = a0lo[i]; a0[i + 8] = a0hi[i];
      a1[i] = a1lo[i]; a1[i + 8] = a1hi[i];
    }
    // B fragments (32x16 f16): lane holds column, K = kb + lh*8 + [0..7], then +16
    v8h b0lo = *(const v8h*)(brow0 + kb);
    v8h b0hi = *(const v8h*)(brow0 + kb + 16);
    v8h b1lo = *(const v8h*)(brow1 + kb);
    v8h b1hi = *(const v8h*)(brow1 + kb + 16);
    v16h b0, b1;
#pragma unroll
    for (int i = 0; i < 8; ++i) {
      b0[i] = b0lo[i]; b0[i + 8] = b0hi[i];
      b1[i] = b1lo[i]; b1[i + 8] = b1hi[i];
    }
    acc00 = __builtin_amdgcn_wmma_f32_16x16x32_f16(false, a0, false, b0, (short)0, acc00, false, false);
    acc01 = __builtin_amdgcn_wmma_f32_16x16x32_f16(false, a0, false, b1, (short)0, acc01, false, false);
    acc10 = __builtin_amdgcn_wmma_f32_16x16x32_f16(false, a1, false, b0, (short)0, acc10, false, false);
    acc11 = __builtin_amdgcn_wmma_f32_16x16x32_f16(false, a1, false, b1, (short)0, acc11, false, false);
  }

  // epilogue: C/D layout -> element (m0 + mi*16 + lh*8 + r, n0 + ni*16 + lane%16)
  const int nc0 = n0 + lr;
  const int nc1 = n0 + 16 + lr;
  const float bv0 = bias[nc0], bv1 = bias[nc1];
  float gv0 = 1.f, bb0 = 0.f, gv1 = 1.f, bb1 = 0.f;
  if (mode == 0) { gv0 = gamma[nc0]; bb0 = beta[nc0]; gv1 = gamma[nc1]; bb1 = beta[nc1]; }
#pragma unroll
  for (int r = 0; r < 8; ++r) {
    const int ma = m0 + lh * 8 + r;
    const int mb = ma + 16;
    float v00 = acc00[r] + bv0, v01 = acc01[r] + bv1;
    float v10 = acc10[r] + bv0, v11 = acc11[r] + bv1;
    if (mode == 0) {
      v00 = v00 * gv0 + bb0; v00 = v00 > 0.f ? v00 : 0.f;
      v01 = v01 * gv1 + bb1; v01 = v01 > 0.f ? v01 : 0.f;
      v10 = v10 * gv0 + bb0; v10 = v10 > 0.f ? v10 : 0.f;
      v11 = v11 * gv1 + bb1; v11 = v11 > 0.f ? v11 : 0.f;
      outH[(size_t)ma * Cout + nc0] = (_Float16)v00;
      outH[(size_t)ma * Cout + nc1] = (_Float16)v01;
      outH[(size_t)mb * Cout + nc0] = (_Float16)v10;
      outH[(size_t)mb * Cout + nc1] = (_Float16)v11;
    } else {
      outF[(size_t)ma * Cout + nc0] = v00;
      outF[(size_t)ma * Cout + nc1] = v01;
      outF[(size_t)mb * Cout + nc0] = v10;
      outF[(size_t)mb * Cout + nc1] = v11;
    }
  }
}

// ---------------- max over the 32 grouped samples ----------------
__global__ void pn_maxpool_kernel(const _Float16* __restrict__ A, _Float16* __restrict__ out,
                                  int rowsOut, int D) {
  int tid = blockIdx.x * blockDim.x + threadIdx.x;
  if (tid >= rowsOut * D) return;
  int s = tid / D, d = tid % D;
  const _Float16* p = A + ((size_t)s * 32) * D + d;
  float m = (float)p[0];
  for (int r = 1; r < 32; ++r) m = fmaxf(m, (float)p[(size_t)r * D]);
  out[tid] = (_Float16)m;
}

// ---------------- host orchestration ----------------
extern "C" void kernel_launch(void* const* d_in, const int* in_sizes, int n_in,
                              void* d_out, int out_size, void* d_ws, size_t ws_size,
                              hipStream_t stream) {
  (void)in_sizes; (void)n_in; (void)out_size; (void)ws_size;

  const float* pc = (const float*)d_in[0];
  auto Wp    = [&](int s, int l) { return (const float*)d_in[1 + (s * 3 + l) * 4 + 0]; };
  auto Bp    = [&](int s, int l) { return (const float*)d_in[1 + (s * 3 + l) * 4 + 1]; };
  auto Gp    = [&](int s, int l) { return (const float*)d_in[1 + (s * 3 + l) * 4 + 2]; };
  auto Betap = [&](int s, int l) { return (const float*)d_in[1 + (s * 3 + l) * 4 + 3]; };
  const float* finalW = (const float*)d_in[37];
  const float* finalB = (const float*)d_in[38];

  char* ws = (char*)d_ws;
  float*     xyz0   = (float*)(ws + OFF_XYZ0);
  float*     nxyzA[3] = { (float*)(ws + OFF_NXYZ1), (float*)(ws + OFF_NXYZ2), (float*)(ws + OFF_NXYZ3) };
  _Float16*  feats0 = (_Float16*)(ws + OFF_FEATS0);
  _Float16*  featsA[3] = { (_Float16*)(ws + OFF_FEATS1), (_Float16*)(ws + OFF_FEATS2), (_Float16*)(ws + OFF_FEATS3) };
  int*       fidx   = (int*)(ws + OFF_FIDX);
  _Float16*  wf16   = (_Float16*)(ws + OFF_WF16);
  _Float16*  bufA   = (_Float16*)(ws + OFF_BUFA);
  _Float16*  bufB   = (_Float16*)(ws + OFF_BUFB);

  const int B = 4, N0 = 8192;
  const int Sarr[3]  = { 2048, 512, 128 };
  const float r2[3]  = { 0.09f, 0.25f, 0.49f };
  const int mlp[3][4] = { {6, 64, 64, 128}, {131, 128, 128, 256}, {259, 256, 256, 256} };
  const int kpad0[3] = { 32, 160, 288 };

  pn_split_kernel<<<(B * N0 + 255) / 256, 256, 0, stream>>>(pc, xyz0, feats0, B * N0);

  const float*    xyz_s   = xyz0;
  const _Float16* feats_s = feats0;
  int Ncur = N0, Cfeat = 3;

  for (int s = 0; s < 3; ++s) {
    const int S = Sarr[s];
    pn_fps_kernel<<<B, FPS_THREADS, 0, stream>>>(xyz_s, fidx, Ncur, S);
    pn_gather_kernel<<<(B * S + 255) / 256, 256, 0, stream>>>(xyz_s, fidx, nxyzA[s], B, Ncur, S);
    pn_group_kernel<<<(B * S) / 8, 256, 0, stream>>>(xyz_s, feats_s, nxyzA[s], bufA,
                                                     B, Ncur, S, Cfeat, kpad0[s], r2[s]);
    const int M = B * S * 32;
    _Float16* aIn = bufA;
    _Float16* aOut = bufB;
    for (int l = 0; l < 3; ++l) {
      const int Cin = mlp[s][l], Cout = mlp[s][l + 1];
      const int Kp = (l == 0) ? kpad0[s] : Cin;       // Cin already multiple of 32 for l>0
      pn_convw_kernel<<<(Cout * Kp + 255) / 256, 256, 0, stream>>>(Wp(s, l), wf16, Cin, Kp, Cout * Kp);
      const int tiles = (M / 32) * (Cout / 32);
      pn_wmma_gemm_kernel<<<(tiles + 7) / 8, 256, 0, stream>>>(
          aIn, wf16, Bp(s, l), Gp(s, l), Betap(s, l), aOut, nullptr, M, Kp, Cout, 0);
      _Float16* t = aIn; aIn = aOut; aOut = t;        // result of layer l now in aIn
    }
    // after 3 layers: results in bufB (A->B, B->A, A->B), aIn == bufB
    pn_maxpool_kernel<<<(B * S * mlp[s][3] + 255) / 256, 256, 0, stream>>>(aIn, featsA[s], B * S, mlp[s][3]);
    xyz_s = nxyzA[s]; feats_s = featsA[s]; Ncur = S; Cfeat = mlp[s][3];
  }

  // final linear: (4*128, 256) x (256, 256)^T + b -> f32 d_out
  pn_convw_kernel<<<(256 * 256 + 255) / 256, 256, 0, stream>>>(finalW, wf16, 256, 256, 256 * 256);
  const int tilesF = (512 / 32) * (256 / 32);
  pn_wmma_gemm_kernel<<<(tilesF + 7) / 8, 256, 0, stream>>>(
      featsA[2], wf16, finalB, nullptr, nullptr, nullptr, (float*)d_out, 512, 256, 256, 1);
}
